// Stage_CLAM_20237885899127
// MI455X (gfx1250) — compile-verified, matrix operand order checked by reading, not run on hardware
//
#include <hip/hip_runtime.h>
#include <hip/hip_bf16.h>

// ---------- types & helpers ----------------------------------------------
typedef __attribute__((ext_vector_type(16))) __bf16 v16bf;
typedef __attribute__((ext_vector_type(8)))  float  v8f;

union BF {
    v16bf v;
    unsigned short u[16];
    uint4 q[2];
};

__device__ __forceinline__ unsigned short f2bf(float f) {
    unsigned int u = __float_as_uint(f);
    unsigned int r = u + 0x7FFFu + ((u >> 16) & 1u);   // round-to-nearest-even
    return (unsigned short)(r >> 16);
}
__device__ __forceinline__ float bf2f(unsigned short s) {
    return __uint_as_float(((unsigned int)s) << 16);
}
// monotonic float<->int encoding for atomicMax-based float max
__device__ __forceinline__ int fenc(float f) {
    int i = __float_as_int(f);
    return i >= 0 ? i : (i ^ 0x7fffffff);
}
__device__ __forceinline__ float fdec(int i) {
    return __int_as_float(i >= 0 ? i : (i ^ 0x7fffffff));
}
__device__ __forceinline__ v8f wmma_bf16(const BF& a, const BF& b, v8f c) {
    // (neg_a, A, neg_b, B, c_mod, C, reuse_a, reuse_b)
    return __builtin_amdgcn_wmma_f32_16x16x32_bf16(false, a.v, false, b.v,
                                                   (short)0, c, false, false);
}
// K index pattern for the 16-bit 16x32 A/B fragment layout (ISA 7.12.2)
__device__ __forceinline__ int kpat(int v, int hf) {
    return (v < 4 ? 2 * v : 16 + 2 * (v - 4)) + 8 * hf;
}

// load 4 consecutive B fragments from fragment-linear packed weights
// layout: [s][t(sT)][lane(32)][e(16)] ushorts; tile stride = 64 uint4
__device__ __forceinline__ void loadB4(BF* b, const unsigned short* __restrict__ base,
                                       int sT, int s, int t0, int lane) {
    const uint4* p = (const uint4*)base + (((size_t)s * sT + t0) * 32 + lane) * 2;
#pragma unroll
    for (int u = 0; u < 4; ++u) {
        b[u].q[0] = p[u * 64];
        b[u].q[1] = p[u * 64 + 1];
    }
}
__device__ __forceinline__ void wmma4(const BF& a, const BF* b, v8f* c) {
#pragma unroll
    for (int u = 0; u < 4; ++u) c[u] = wmma_bf16(a, b[u], c[u]);
}

// ---------- kernel 0: pack W1/Wa/Wb to fragment-linear bf16 ---------------
__global__ void clam_pack(const float* __restrict__ W1,
                          const float* __restrict__ Wa,
                          const float* __restrict__ Wb,
                          unsigned short* __restrict__ w1p,
                          unsigned short* __restrict__ wap,
                          unsigned short* __restrict__ wbp,
                          int* __restrict__ maxenc) {
    int idx = blockIdx.x * 256 + threadIdx.x;
    if (idx < 2) maxenc[idx] = fenc(-__builtin_inff());
    const int T1 = 32 * 32 * 32 * 16;          // 524288
    const int T2 = 16 * 16 * 32 * 16;          // 131072
    if (idx < T1) {
        int e = idx & 15, ln = (idx >> 4) & 31, t = (idx >> 9) & 31, s = idx >> 14;
        int n = t * 16 + (ln & 15);
        int k = s * 32 + kpat(e >> 1, ln >> 4) + (e & 1);
        w1p[idx] = f2bf(W1[n * 1024 + k]);
    } else if (idx < T1 + 2 * T2) {
        int j = idx - T1;
        const float* Wsrc = Wa;
        unsigned short* Wdst = wap;
        if (j >= T2) { j -= T2; Wsrc = Wb; Wdst = wbp; }
        int e = j & 15, ln = (j >> 4) & 31, t = (j >> 9) & 15, s = j >> 13;
        int n = t * 16 + (ln & 15);
        int k = s * 32 + kpat(e >> 1, ln >> 4) + (e & 1);
        Wdst[j] = f2bf(Wsrc[n * 512 + k]);
    }
}

// ---------- kernel 1: fused GEMM1 + GEMM2 + attention logits --------------
// block = 128 threads (4 waves). 32 rows of h per block.
__global__ __launch_bounds__(128) void clam_main(
        const float* __restrict__ h,  const float* __restrict__ b1,
        const float* __restrict__ ba, const float* __restrict__ bb,
        const float* __restrict__ Wc, const float* __restrict__ bc,
        const unsigned short* __restrict__ w1p,
        const unsigned short* __restrict__ wap,
        const unsigned short* __restrict__ wbp,
        unsigned short* __restrict__ xbf, float* __restrict__ logA, int Nn) {
    __shared__ unsigned short hs[32][40];    // h tile, padded stride (no conflicts)
    __shared__ unsigned short xs[32][520];   // relu(x) tile in bf16
    __shared__ float red[32][4];             // [row][class*2 + wc] partial logits

    const int tid  = threadIdx.x;
    const int lane = tid & 31;
    const int wave = tid >> 5;
    const int wr = wave >> 1, wc = wave & 1;
    const int hf = lane >> 4;                // lane half
    const int lm = lane & 15;
    const int row0 = blockIdx.x * 32;

    // ---------------- GEMM1: x = relu(h @ W1^T + b1), 32x512 tile --------
    v8f acc[16];
#pragma unroll
    for (int t = 0; t < 16; ++t)
#pragma unroll
        for (int i = 0; i < 8; ++i) acc[t][i] = 0.0f;

    BF bA[4], bB[4];
    loadB4(bA, w1p, 32, 0, wc * 16 + 0, lane);   // prime the pipeline

    for (int s = 0; s < 32; ++s) {
        __syncthreads();
        // stage h tile 32x32 fp32 -> bf16 (vectorized: float4 in, dword out)
#pragma unroll
        for (int ii = 0; ii < 2; ++ii) {
            int idx = tid + ii * 128;            // 256 quads of 4 floats
            int r = idx >> 3, kq = (idx & 7) * 4;
            int grow = row0 + r; if (grow >= Nn) grow = Nn - 1;
            const float4 f = *(const float4*)&h[(size_t)grow * 1024 + s * 32 + kq];
            *(unsigned int*)&hs[r][kq]     = (unsigned int)f2bf(f.x) | ((unsigned int)f2bf(f.y) << 16);
            *(unsigned int*)&hs[r][kq + 2] = (unsigned int)f2bf(f.z) | ((unsigned int)f2bf(f.w) << 16);
        }
        if (s + 1 < 32) {  // gfx1250 global_prefetch of next h tile
            int r = tid >> 3, kq = (tid & 7) * 4;
            int grow = row0 + r; if (grow >= Nn) grow = Nn - 1;
            __builtin_prefetch(&h[(size_t)grow * 1024 + (s + 1) * 32 + kq], 0, 1);
        }
        __syncthreads();

        BF af;  // A fragment: rows 16*wr..+15 of h tile
        {
            int m = 16 * wr + lm;
#pragma unroll
            for (int v = 0; v < 8; ++v) {
                int kb = kpat(v, hf);
                af.u[2 * v]     = hs[m][kb];
                af.u[2 * v + 1] = hs[m][kb + 1];
            }
        }
        // software-pipelined: load group g+1 while matrix pipe runs group g
        loadB4(bB, w1p, 32, s, wc * 16 + 4, lane);
        wmma4(af, bA, &acc[0]);
        loadB4(bA, w1p, 32, s, wc * 16 + 8, lane);
        wmma4(af, bB, &acc[4]);
        loadB4(bB, w1p, 32, s, wc * 16 + 12, lane);
        wmma4(af, bA, &acc[8]);
        if (s + 1 < 32) loadB4(bA, w1p, 32, s + 1, wc * 16 + 0, lane);
        wmma4(af, bB, &acc[12]);
    }
    __syncthreads();

    // epilogue: bias + relu -> xs (bf16)
#pragma unroll
    for (int t = 0; t < 16; ++t) {
        int col = wc * 256 + t * 16 + lm;
        float bias = b1[col];
#pragma unroll
        for (int i = 0; i < 8; ++i) {
            int m = 16 * wr + i + 8 * hf;
            float v = acc[t][i] + bias;
            xs[m][col] = f2bf(v > 0.0f ? v : 0.0f);
        }
    }
    __syncthreads();

    // spill x tile to global (bf16, coalesced dwords) for the A@x pass
    for (int i = tid; i < 32 * 256; i += 128) {
        int r = i >> 8, cu = i & 255;
        unsigned int val = *(const unsigned int*)&xs[r][cu * 2];
        ((unsigned int*)xbf)[((size_t)(row0 + r)) * 256 + cu] = val;
    }

    // ---------------- GEMM2: a_pre / g_pre, 32x256 each -------------------
    v8f aacc[8], gacc[8];
#pragma unroll
    for (int t = 0; t < 8; ++t)
#pragma unroll
        for (int i = 0; i < 8; ++i) { aacc[t][i] = 0.0f; gacc[t][i] = 0.0f; }

    BF cA[4], cB[4];
    loadB4(cA, wap, 16, 0, wc * 8 + 0, lane);

    for (int s = 0; s < 16; ++s) {
        BF xf;
        {
            int m = 16 * wr + lm;
#pragma unroll
            for (int v = 0; v < 8; ++v) {
                int kb = s * 32 + kpat(v, hf);
                xf.u[2 * v]     = xs[m][kb];
                xf.u[2 * v + 1] = xs[m][kb + 1];
            }
        }
        loadB4(cB, wbp, 16, s, wc * 8 + 0, lane);
        wmma4(xf, cA, &aacc[0]);
        loadB4(cA, wap, 16, s, wc * 8 + 4, lane);
        wmma4(xf, cB, &gacc[0]);
        loadB4(cB, wbp, 16, s, wc * 8 + 4, lane);
        wmma4(xf, cA, &aacc[4]);
        if (s + 1 < 16) loadB4(cA, wap, 16, s + 1, wc * 8 + 0, lane);
        wmma4(xf, cB, &gacc[4]);
    }

    // epilogue: tanh * sigmoid, project onto Wc (N_CLASSES=2) per row
    float pr0[8], pr1[8];
#pragma unroll
    for (int i = 0; i < 8; ++i) { pr0[i] = 0.0f; pr1[i] = 0.0f; }
#pragma unroll
    for (int t = 0; t < 8; ++t) {
        int col = (wc * 8 + t) * 16 + lm;
        float bav = ba[col], bbv = bb[col];
        float w0 = Wc[col], w1v = Wc[256 + col];
#pragma unroll
        for (int i = 0; i < 8; ++i) {
            float av = tanhf(aacc[t][i] + bav);
            float gv = 1.0f / (1.0f + __expf(-(gacc[t][i] + bbv)));
            float tv = av * gv;
            pr0[i] += tv * w0;
            pr1[i] += tv * w1v;
        }
    }
#pragma unroll
    for (int i = 0; i < 8; ++i) {
        float s0 = pr0[i], s1 = pr1[i];
#pragma unroll
        for (int off = 1; off < 16; off <<= 1) {
            s0 += __shfl_xor(s0, off, 16);
            s1 += __shfl_xor(s1, off, 16);
        }
        if (lm == 0) {  // lanes 0 and 16: unique (row, wc) slots -> deterministic
            int m = 16 * wr + i + 8 * hf;
            red[m][0 * 2 + wc] = s0;
            red[m][1 * 2 + wc] = s1;
        }
    }
    __syncthreads();
    if (tid < 64) {
        int m = tid >> 1, c = tid & 1;
        int grow = row0 + m;
        if (grow < Nn)
            logA[(size_t)c * Nn + grow] = red[m][c * 2] + red[m][c * 2 + 1] + bc[c];
    }
}

// ---------- kernel 2: per-class max over instances ------------------------
__global__ void clam_max(const float* __restrict__ logA, int Nn,
                         int* __restrict__ maxenc) {
    __shared__ float s0[256], s1[256];
    int tid = threadIdx.x;
    float m0 = -__builtin_inff(), m1 = -__builtin_inff();
    for (int i = blockIdx.x * 256 + tid; i < Nn; i += gridDim.x * 256) {
        m0 = fmaxf(m0, logA[i]);
        m1 = fmaxf(m1, logA[Nn + i]);
    }
    s0[tid] = m0; s1[tid] = m1;
    __syncthreads();
    for (int st = 128; st > 0; st >>= 1) {
        if (tid < st) {
            s0[tid] = fmaxf(s0[tid], s0[tid + st]);
            s1[tid] = fmaxf(s1[tid], s1[tid + st]);
        }
        __syncthreads();
    }
    if (tid == 0) {
        atomicMax(&maxenc[0], fenc(s0[0]));
        atomicMax(&maxenc[1], fenc(s1[0]));
    }
}

// ---------- kernel 3: e = exp(logit-max), block partials of sum and A@x ---
__global__ __launch_bounds__(256) void clam_expsum(
        const float* __restrict__ logA, const unsigned short* __restrict__ xbf,
        const int* __restrict__ maxenc, float* __restrict__ outA,
        float* __restrict__ Mpart, float* __restrict__ esum_part, int Nn) {
    __shared__ float es0[128], es1[128];
    int tid = threadIdx.x;
    int row0 = blockIdx.x * 128;
    float mx0 = fdec(maxenc[0]), mx1 = fdec(maxenc[1]);
    {
        int r = tid >> 1, c = tid & 1;
        int grow = row0 + r;
        float e = 0.0f;
        if (grow < Nn) {
            float l = logA[(size_t)c * Nn + grow];
            e = __expf(l - (c ? mx1 : mx0));
            outA[2 + (size_t)c * Nn + grow] = e;  // unnormalized, fixed later
        }
        if (c) es1[r] = e; else es0[r] = e;
    }
    __syncthreads();
    if (tid < 2) {
        float s = 0.0f;
        for (int r = 0; r < 128; ++r) s += (tid ? es1[r] : es0[r]);
        esum_part[blockIdx.x * 2 + tid] = s;
    }
    // partial M[c][j] = sum_r e[c][r] * x[r][j]  (x read as dwords: 2 bf16)
    float a00 = 0, a01 = 0, a10 = 0, a11 = 0;
    const unsigned int* xr = (const unsigned int*)(xbf + (size_t)row0 * 512);
    for (int r = 0; r < 128; ++r) {
        float e0 = es0[r], e1 = es1[r];
        unsigned int w = xr[(size_t)r * 256 + tid];
        float x0 = bf2f((unsigned short)(w & 0xFFFFu));
        float x1 = bf2f((unsigned short)(w >> 16));
        a00 += e0 * x0; a10 += e1 * x0;
        a01 += e0 * x1; a11 += e1 * x1;
    }
    float* mp = Mpart + (size_t)blockIdx.x * 1024;
    int j0 = 2 * tid, j1 = 2 * tid + 1;
    mp[j0] = a00; mp[j1] = a01; mp[512 + j0] = a10; mp[512 + j1] = a11;
}

// ---------- kernel 4: deterministic reduction of partials -----------------
__global__ void clam_reduce(const float* __restrict__ Mpart,
                            const float* __restrict__ esum_part,
                            float* __restrict__ Macc, float* __restrict__ sumv,
                            int nb) {
    int g = blockIdx.x * 256 + threadIdx.x;
    if (g < 1024) {
        float a = 0.0f;
        for (int b = 0; b < nb; ++b) a += Mpart[(size_t)b * 1024 + g];
        Macc[g] = a;
    } else if (g < 1026) {
        int c = g - 1024;
        float s = 0.0f;
        for (int b = 0; b < nb; ++b) s += esum_part[b * 2 + c];
        sumv[c] = s;
    }
}

// ---------- kernel 5: normalize A --------------------------------------
__global__ void clam_norm(float* __restrict__ outA,
                          const float* __restrict__ sumv, int Nn) {
    int i = blockIdx.x * 256 + threadIdx.x;
    if (i < 2 * Nn) {
        int c = i >= Nn;
        outA[2 + i] = outA[2 + i] / sumv[c];
    }
}

// ---------- kernel 6: bag logits -----------------------------------------
__global__ void clam_logits(const float* __restrict__ Macc,
                            const float* __restrict__ sumv,
                            const float* __restrict__ Wcls,
                            const float* __restrict__ bcls,
                            float* __restrict__ out) {
    __shared__ float sm[128];
    int tid = threadIdx.x;
    int c = tid >> 6, jl = tid & 63;
    float p = 0.0f;
    for (int j = jl; j < 512; j += 64) p += Macc[c * 512 + j] * Wcls[c * 512 + j];
    sm[tid] = p;
    __syncthreads();
    if ((tid & 63) == 0) {
        float s = 0.0f;
        for (int k = 0; k < 64; ++k) s += sm[c * 64 + k];
        out[c] = s / sumv[c] + bcls[c];
    }
}

// ---------- host launch ---------------------------------------------------
extern "C" void kernel_launch(void* const* d_in, const int* in_sizes, int n_in,
                              void* d_out, int out_size, void* d_ws, size_t ws_size,
                              hipStream_t stream) {
    const float* h    = (const float*)d_in[0];
    const float* W1   = (const float*)d_in[1];
    const float* b1   = (const float*)d_in[2];
    const float* Wa   = (const float*)d_in[3];
    const float* ba   = (const float*)d_in[4];
    const float* Wb   = (const float*)d_in[5];
    const float* bb   = (const float*)d_in[6];
    const float* Wc   = (const float*)d_in[7];
    const float* bc   = (const float*)d_in[8];
    const float* Wcls = (const float*)d_in[9];
    const float* bcls = (const float*)d_in[10];
    (void)n_in; (void)out_size; (void)ws_size;

    const int Nn = in_sizes[0] / 1024;             // 150000
    const int nblk_main = (Nn + 31) / 32;          // 4688
    const int NPAD = nblk_main * 32;               // 150016
    const int nb4 = NPAD / 128;                    // 1172

    char* ws = (char*)d_ws;
    size_t off = 0;
    auto take = [&](size_t bytes) {
        char* p = ws + off;
        off = (off + bytes + 255) & ~(size_t)255;
        return p;
    };
    unsigned short* w1p  = (unsigned short*)take((size_t)512 * 1024 * 2);
    unsigned short* wap  = (unsigned short*)take((size_t)256 * 512 * 2);
    unsigned short* wbp  = (unsigned short*)take((size_t)256 * 512 * 2);
    unsigned short* xbf  = (unsigned short*)take((size_t)NPAD * 512 * 2);
    float* logA          = (float*)take((size_t)2 * Nn * 4);
    int*   maxenc        = (int*)take(8);
    float* sumv          = (float*)take(8);
    float* Macc          = (float*)take(1024 * 4);
    float* Mpart         = (float*)take((size_t)nb4 * 1024 * 4);
    float* esum          = (float*)take((size_t)nb4 * 2 * 4);

    float* out = (float*)d_out;

    clam_pack  <<<3072, 256, 0, stream>>>(W1, Wa, Wb, w1p, wap, wbp, maxenc);
    clam_main  <<<nblk_main, 128, 0, stream>>>(h, b1, ba, bb, Wc, bc,
                                               w1p, wap, wbp, xbf, logA, Nn);
    clam_max   <<<512, 256, 0, stream>>>(logA, Nn, maxenc);
    clam_expsum<<<nb4, 256, 0, stream>>>(logA, xbf, maxenc, out, Mpart, esum, Nn);
    clam_reduce<<<5, 256, 0, stream>>>(Mpart, esum, Macc, sumv, nb4);
    clam_norm  <<<(2 * Nn + 255) / 256, 256, 0, stream>>>(out, sumv, Nn);
    clam_logits<<<1, 128, 0, stream>>>(Macc, sumv, Wcls, bcls, out);
}
